// WaveClipEncoder_84439057039675
// MI455X (gfx1250) — matrix operand order repaired and checked by reading, not compiled, not run on hardware
//
#include <hip/hip_runtime.h>
#include <math.h>

typedef __attribute__((ext_vector_type(16))) __bf16 v16bf;
typedef __attribute__((ext_vector_type(8)))  float  v8f;
typedef unsigned short ushort_t;
typedef unsigned int   uint_t;

union ABfrag { uint4 q[2]; v16bf v; };

// ---------- constants ----------
#define NB      64
#define CCH     3
#define HH      224
#define WW      224
#define HW      (HH*WW)          // 50176
#define PP      14
#define POS     (PP*PP)          // 196
#define D540    540
#define CIN     30
#define NT_N    34               // 544/16
#define K1PAD   32
#define K2PAD   544              // 17 tiles of 32
#define NT_K2   17
#define GICH    1080
#define GSTR    1088             // gi channel stride (34 tiles of 32)
#define NT_KC   34
#define MROWS   (NB*POS)         // 12544
#define MT_ALL  (MROWS/16)       // 784
#define MT_IMG  13               // ceil(196/16)
#define QK0     49673            // floor(0.99*(50176-1))
#define QFRAC   0.25f
#define EPSQ    1e-4f

__device__ __forceinline__ ushort_t f2bf(float f) {
  uint_t u = __float_as_uint(f);
  u += 0x7FFFu + ((u >> 16) & 1u);
  return (ushort_t)(u >> 16);
}

// Haar a-trous band at (y,x): t=0 cH, 1 cV, 2 cD, 3 cA
__device__ __forceinline__ float haar_band(const float* ap, int y, int x, int d, int t) {
  int xd = x + d; if (xd >= WW) xd -= WW;
  int yd = y + d; if (yd >= HH) yd -= HH;
  float a00 = ap[y*WW+x],  a01 = ap[y*WW+xd];
  float a10 = ap[yd*WW+x], a11 = ap[yd*WW+xd];
  float sx0 = a00 + a01, dx0 = a00 - a01;
  float sx1 = a10 + a11, dx1 = a10 - a11;
  if (t == 0) return 0.5f*(sx0 - sx1);
  if (t == 1) return 0.5f*(dx0 + dx1);
  if (t == 2) return 0.5f*(dx0 - dx1);
  return 0.5f*(sx0 + sx1);
}

// Exact k-th order statistic of |band| (or |a-mu| for LL) via 3-pass radix
// histogram over the 31-bit non-negative float pattern (order-preserving).
__device__ float block_select_kth(const float* ap, int d, int t, float mu, int isLL,
                                  int* hist, volatile uint_t* s_prefix,
                                  volatile int* s_rank, int k) {
  const int tid = threadIdx.x, nth = blockDim.x;
  if (tid == 0) { *s_prefix = 0u; *s_rank = k; }
  __syncthreads();
  const int shifts[3] = {20, 9, 0};
  const int bitsv[3]  = {11, 11, 9};
  for (int p = 0; p < 3; ++p) {
    const int nb = 1 << bitsv[p];
    for (int i = tid; i < nb; i += nth) hist[i] = 0;
    __syncthreads();
    const uint_t pref = *s_prefix;
    const int sh = shifts[p];
    const uint_t hiSh = (uint_t)(sh + bitsv[p]);
    for (int idx = tid; idx < HW; idx += nth) {
      float v;
      if (isLL) v = fabsf(ap[idx] - mu);
      else      v = fabsf(haar_band(ap, idx/WW, idx%WW, d, t));
      uint_t key = __float_as_uint(v);
      if ((key >> hiSh) == (pref >> hiSh))
        atomicAdd(&hist[(key >> sh) & (nb - 1)], 1);
    }
    __syncthreads();
    if (tid == 0) {
      int r = *s_rank, cum = 0, b = 0;
      for (; b < nb; ++b) { int c = hist[b]; if (r < cum + c) break; cum += c; }
      if (b >= nb) b = nb - 1;
      *s_rank = r - cum;
      *s_prefix = pref | ((uint_t)b << sh);
    }
    __syncthreads();
  }
  float res = __uint_as_float(*s_prefix);
  __syncthreads();
  return res;
}

// One SWT level: write cA plane, and for each detail band do exact q99 + pool.
__global__ void swt_level_kernel(const float* __restrict__ a_in, float* __restrict__ a_out,
                                 float* __restrict__ e_pix, int d, int jlev) {
  const int plane = blockIdx.x;                 // n*3+c
  const float* ap = a_in + (size_t)plane * HW;
  float* aout = a_out + (size_t)plane * HW;
  const int n = plane / CCH, c = plane % CCH;
  __shared__ int hist[2048];
  __shared__ uint_t s_prefix; __shared__ int s_rank;
  const int tid = threadIdx.x, nth = blockDim.x;

  for (int idx = tid; idx < HW; idx += nth)
    aout[idx] = haar_band(ap, idx/WW, idx%WW, d, 3);

  for (int t = 0; t < 3; ++t) {
    float v0 = block_select_kth(ap, d, t, 0.f, 0, hist, &s_prefix, &s_rank, QK0);
    float v1 = block_select_kth(ap, d, t, 0.f, 0, hist, &s_prefix, &s_rank, QK0 + 1);
    float q = v0 + QFRAC * (v1 - v0);
    float inv = 1.0f / (q + EPSQ);
    int ch = 3*(3*jlev + t) + c;
    float* eout = e_pix + ((size_t)n*CIN + ch) * POS;
    for (int cell = tid; cell < POS; cell += nth) {
      int py = cell / PP, px = cell % PP;
      float s = 0.f;
      for (int iy = 0; iy < 16; ++iy)
        for (int ix = 0; ix < 16; ++ix)
          s += haar_band(ap, py*16+iy, px*16+ix, d, t);
      eout[cell] = s * (1.0f/256.0f) * inv;
    }
    __syncthreads();
  }
}

// LL band: mean-center, q99 of |a-mu|, pool.
__global__ void ll_kernel(const float* __restrict__ a_in, float* __restrict__ e_pix) {
  const int plane = blockIdx.x;
  const float* ap = a_in + (size_t)plane * HW;
  const int n = plane / CCH, c = plane % CCH;
  __shared__ int hist[2048];
  __shared__ float sred[256];
  __shared__ uint_t s_prefix; __shared__ int s_rank; __shared__ float s_mu;
  const int tid = threadIdx.x, nth = blockDim.x;

  float part = 0.f;
  for (int idx = tid; idx < HW; idx += nth) part += ap[idx];
  sred[tid] = part; __syncthreads();
  for (int s = nth >> 1; s > 0; s >>= 1) {
    if (tid < s) sred[tid] += sred[tid + s];
    __syncthreads();
  }
  if (tid == 0) s_mu = sred[0] / (float)HW;
  __syncthreads();
  const float mu = s_mu;

  float v0 = block_select_kth(ap, 0, 0, mu, 1, hist, &s_prefix, &s_rank, QK0);
  float v1 = block_select_kth(ap, 0, 0, mu, 1, hist, &s_prefix, &s_rank, QK0 + 1);
  float q = v0 + QFRAC * (v1 - v0);
  float inv = 1.0f / (q + EPSQ);
  float* eout = e_pix + ((size_t)n*CIN + 27 + c) * POS;
  for (int cell = tid; cell < POS; cell += nth) {
    int py = cell / PP, px = cell % PP;
    float s = 0.f;
    for (int iy = 0; iy < 16; ++iy)
      for (int ix = 0; ix < 16; ++ix)
        s += ap[(py*16+iy)*WW + (px*16+ix)];
    eout[cell] = (s * (1.0f/256.0f) - mu) * inv;
  }
}

// ---------- packing kernels ----------
// gi layout: [n][pos (196)][ch (1088)] bf16; channels 540..1079 written by gemm2,
// 1080..1087 zero-filled, 0..539 = t_clip.
__global__ void pack_tclip_kernel(const float* __restrict__ t_clip, ushort_t* __restrict__ gi) {
  int i = blockIdx.x * blockDim.x + threadIdx.x;
  if (i >= NB*D540*POS) return;
  int n = i / (D540*POS);
  int rem = i - n * (D540*POS);
  int ch = rem / POS, pos = rem % POS;
  gi[((size_t)n*POS + pos)*GSTR + ch] = f2bf(t_clip[i]);
}

__global__ void zerofill_gi_kernel(ushort_t* __restrict__ gi) {
  int i = blockIdx.x * blockDim.x + threadIdx.x;
  if (i >= NB*POS*8) return;
  int n = i / (POS*8);
  int r = i % (POS*8);
  int pos = r / 8, ch = GICH + (r % 8);
  gi[((size_t)n*POS + pos)*GSTR + ch] = 0;
}

__global__ void pack_a1_kernel(const float* __restrict__ e_pix, ushort_t* __restrict__ A1) {
  int i = blockIdx.x * blockDim.x + threadIdx.x;
  if (i >= MROWS * K1PAD) return;
  int row = i / K1PAD, cc = i % K1PAD;
  int n = row / POS, pos = row % POS;
  float v = (cc < CIN) ? e_pix[((size_t)n*CIN + cc)*POS + pos] : 0.f;
  A1[i] = f2bf(v);
}

// Fragment-order packing: within a 16x16x32 B tile, element (lane,j) holds
// B[k = 16*(lane>>4) + j][n = lane&15]  (B laid out K x N).
__global__ void pack_w1_kernel(const float* __restrict__ w1, ushort_t* __restrict__ w1f) {
  int i = blockIdx.x * blockDim.x + threadIdx.x;
  if (i >= NT_N * 512) return;
  int nt = i / 512, l = (i % 512) / 16, j = i % 16;
  int oc = nt*16 + (l & 15);
  int k  = 16*(l >> 4) + j;
  float v = (oc < D540 && k < CIN) ? w1[(size_t)oc*CIN + k] : 0.f;
  w1f[i] = f2bf(v);
}

__global__ void pack_w2_kernel(const float* __restrict__ w2, ushort_t* __restrict__ w2f) {
  int i = blockIdx.x * blockDim.x + threadIdx.x;
  if (i >= NT_K2 * NT_N * 512) return;
  int frag = i / 512, l = (i % 512) / 16, j = i % 16;
  int kt = frag / NT_N, nt = frag % NT_N;
  int oc = nt*16 + (l & 15);
  int k  = kt*32 + 16*(l >> 4) + j;
  float v = (oc < D540 && k < D540) ? w2[(size_t)oc*D540 + k] : 0.f;
  w2f[i] = f2bf(v);
}

__global__ void pack_wg_kernel(const float* __restrict__ wg, ushort_t* __restrict__ wgf) {
  int i = blockIdx.x * blockDim.x + threadIdx.x;
  if (i >= 9 * NT_KC * NT_N * 512) return;
  int frag = i / 512, l = (i % 512) / 16, j = i % 16;
  int t  = frag / (NT_KC * NT_N);
  int r2 = frag % (NT_KC * NT_N);
  int kt = r2 / NT_N, nt = r2 % NT_N;
  int oc = nt*16 + (l & 15);
  int ic = kt*32 + 16*(l >> 4) + j;
  int ky = t / 3, kx = t % 3;
  float v = (oc < D540 && ic < GICH)
          ? wg[(((size_t)oc*GICH + ic)*3 + ky)*3 + kx] : 0.f;
  wgf[i] = f2bf(v);
}

// ---------- GEMM1: h = relu(e_pix * w1^T + b1), M=12544 K=32 N=544 ----------
__global__ void gemm1_kernel(const ushort_t* __restrict__ A1, const ushort_t* __restrict__ w1f,
                             const float* __restrict__ b1, ushort_t* __restrict__ h_bf) {
  int wave = (blockIdx.x * blockDim.x + threadIdx.x) >> 5;
  int lane = threadIdx.x & 31;
  if (wave >= MT_ALL * NT_N) return;
  int mt = wave / NT_N, nt = wave % NT_N;
  int arow = mt*16 + (lane & 15);
  int c1 = (lane < 16) ? 0 : 16;
  ABfrag a, b;
  a.q[0] = *(const uint4*)(A1 + (size_t)arow*K1PAD + c1);
  a.q[1] = *(const uint4*)(A1 + (size_t)arow*K1PAD + c1 + 8);
  const ushort_t* bp = w1f + (size_t)nt*512 + lane*16;
  b.q[0] = *(const uint4*)(bp);
  b.q[1] = *(const uint4*)(bp + 8);
  v8f acc = {0.f,0.f,0.f,0.f,0.f,0.f,0.f,0.f};
  acc = __builtin_amdgcn_wmma_f32_16x16x32_bf16(false, a.v, false, b.v, (short)0, acc, false, false);
  int oc = nt*16 + (lane & 15);
  float bias = (oc < D540) ? b1[oc] : 0.f;
#pragma unroll
  for (int r = 0; r < 8; ++r) {
    int M = r + 8*(lane >> 4);
    int row = mt*16 + M;
    float h = acc[r] + bias;
    h = h > 0.f ? h : 0.f;
    h_bf[(size_t)row*K2PAD + oc] = f2bf(h);
  }
}

// ---------- GEMM2: m = tanh(h * w2^T + b2), M=12544 K=544 N=544 ----------
__global__ void gemm2_kernel(const ushort_t* __restrict__ h_bf, const ushort_t* __restrict__ w2f,
                             const float* __restrict__ b2, float* __restrict__ m_f32,
                             ushort_t* __restrict__ gi) {
  int wave = (blockIdx.x * blockDim.x + threadIdx.x) >> 5;
  int lane = threadIdx.x & 31;
  if (wave >= MT_ALL * 9) return;
  int mt = wave / 9, ntg = wave % 9;
  int arow = mt*16 + (lane & 15);
  v8f acc[4];
#pragma unroll
  for (int i = 0; i < 4; ++i) acc[i] = v8f{0.f,0.f,0.f,0.f,0.f,0.f,0.f,0.f};
  for (int kt = 0; kt < NT_K2; ++kt) {
    int c1 = kt*32 + ((lane < 16) ? 0 : 16);
    ABfrag a;
    a.q[0] = *(const uint4*)(h_bf + (size_t)arow*K2PAD + c1);
    a.q[1] = *(const uint4*)(h_bf + (size_t)arow*K2PAD + c1 + 8);
#pragma unroll
    for (int i = 0; i < 4; ++i) {
      int nt = ntg*4 + i;
      if (nt < NT_N) {
        ABfrag b;
        const ushort_t* bp = w2f + ((size_t)(kt*NT_N + nt))*512 + lane*16;
        b.q[0] = *(const uint4*)bp;
        b.q[1] = *(const uint4*)(bp + 8);
        acc[i] = __builtin_amdgcn_wmma_f32_16x16x32_bf16(false, a.v, false, b.v, (short)0, acc[i], false, false);
      }
    }
  }
#pragma unroll
  for (int i = 0; i < 4; ++i) {
    int nt = ntg*4 + i;
    if (nt >= NT_N) continue;
    int oc = nt*16 + (lane & 15);
    if (oc >= D540) continue;
    float bias = b2[oc];
#pragma unroll
    for (int r = 0; r < 8; ++r) {
      int M = r + 8*(lane >> 4);
      int row = mt*16 + M;
      int n = row / POS, pos = row % POS;
      float val = tanhf(acc[i][r] + bias);
      m_f32[((size_t)n*D540 + oc)*POS + pos] = val;
      gi[((size_t)n*POS + pos)*GSTR + D540 + oc] = f2bf(val);
    }
  }
}

// ---------- implicit-GEMM 3x3 SAME conv + gated fusion ----------
// gi tile [pos][32ch] streams into double-buffered LDS via async copies
// (global_load_async_to_lds_b128 + s_wait_asynccnt) overlapped with WMMA.
__global__ void __launch_bounds__(128)
conv_kernel(const ushort_t* __restrict__ gi, const ushort_t* __restrict__ wgf,
            const float* __restrict__ bg, const float* __restrict__ t_clip,
            const float* __restrict__ m_f32, float* __restrict__ out) {
  const int n = blockIdx.x / MT_IMG, mt = blockIdx.x % MT_IMG;
  const int tid = threadIdx.x, lane = tid & 31, wave = tid >> 5;
  __shared__ __align__(16) ushort_t giB[2][197*32];   // [pos][ch], row 196 = zeros
  if (tid < 32) { giB[0][196*32 + tid] = 0; giB[1][196*32 + tid] = 0; }

  const ushort_t* gbase = gi + (size_t)n * POS * GSTR;   // [pos][1088]
  const uint_t lds0 = (uint_t)(uintptr_t)(&giB[0][0]);
  const uint_t lds1 = (uint_t)(uintptr_t)(&giB[1][0]);

  // prefetch kt = 0 into buffer 0: 196 rows x 4 chunks of 16B
  for (int c = tid; c < POS*4; c += 128) {
    int row = c >> 2, cw = c & 3;
    uint_t ldst = lds0 + (uint_t)(row*64 + cw*16);
    const void* g = (const char*)gbase + (size_t)row*(GSTR*2) + cw*16;
    asm volatile("global_load_async_to_lds_b128 %0, %1, off" :: "v"(ldst), "v"(g) : "memory");
  }

  v8f acc[9];
#pragma unroll
  for (int j = 0; j < 9; ++j) acc[j] = v8f{0.f,0.f,0.f,0.f,0.f,0.f,0.f,0.f};

  const int m = lane & 15;
  const int pos = mt*16 + m;
  const int y = pos / PP, x = pos % PP;
  const int ch1 = (lane < 16) ? 0 : 16;

  for (int kt = 0; kt < NT_KC; ++kt) {
    asm volatile("s_wait_asynccnt 0x0" ::: "memory");
    __syncthreads();                       // tile kt resident in giB[kt&1]
    const int cur = kt & 1;
    if (kt + 1 < NT_KC) {                  // stream next tile into other buffer
      const uint_t ldsn = cur ? lds0 : lds1;
      const char* src = (const char*)(gbase + (kt + 1)*32);
      for (int c = tid; c < POS*4; c += 128) {
        int row = c >> 2, cw = c & 3;
        uint_t ldst = ldsn + (uint_t)(row*64 + cw*16);
        const void* g = src + (size_t)row*(GSTR*2) + cw*16;
        asm volatile("global_load_async_to_lds_b128 %0, %1, off" :: "v"(ldst), "v"(g) : "memory");
      }
    }
    const ushort_t* lcur = &giB[cur][0];
#pragma unroll
    for (int t = 0; t < 9; ++t) {
      const int ky = t/3 - 1, kx = t%3 - 1;
      const int sy = y + ky, sx = x + kx;
      const int rrow = (pos < POS && sy >= 0 && sy < PP && sx >= 0 && sx < PP)
                     ? (sy*PP + sx) : 196;
      ABfrag a;
      a.q[0] = *(const uint4*)(lcur + rrow*32 + ch1);
      a.q[1] = *(const uint4*)(lcur + rrow*32 + ch1 + 8);
#pragma unroll
      for (int j = 0; j < 9; ++j) {
        int nt = wave + 4*j;
        if (nt < NT_N) {
          ABfrag b;
          const ushort_t* bp = wgf + ((size_t)((t*NT_KC + kt)*NT_N + nt))*512 + lane*16;
          b.q[0] = *(const uint4*)bp;
          b.q[1] = *(const uint4*)(bp + 8);
          acc[j] = __builtin_amdgcn_wmma_f32_16x16x32_bf16(false, a.v, false, b.v, (short)0, acc[j], false, false);
        }
      }
    }
  }
#pragma unroll
  for (int j = 0; j < 9; ++j) {
    int nt = wave + 4*j;
    if (nt >= NT_N) continue;
    int oc = nt*16 + (lane & 15);
    if (oc >= D540) continue;
    float bias = bg[oc];
#pragma unroll
    for (int r = 0; r < 8; ++r) {
      int M = r + 8*(lane >> 4);
      int p2 = mt*16 + M;
      if (p2 < POS) {
        size_t idx = ((size_t)n*D540 + oc)*POS + p2;
        float g = acc[j][r] + bias;
        float gamma = 1.f / (1.f + expf(-g));
        out[idx] = t_clip[idx] + gamma * m_f32[idx];
      }
    }
  }
}

// ---------- host ----------
static inline size_t alignup(size_t v) { return (v + 255) & ~(size_t)255; }

extern "C" void kernel_launch(void* const* d_in, const int* in_sizes, int n_in,
                              void* d_out, int out_size, void* d_ws, size_t ws_size,
                              hipStream_t stream) {
  (void)in_sizes; (void)n_in; (void)out_size; (void)ws_size;
  const float* frames = (const float*)d_in[0];
  const float* t_clip = (const float*)d_in[1];
  const float* w1     = (const float*)d_in[2];
  const float* b1     = (const float*)d_in[3];
  const float* w2     = (const float*)d_in[4];
  const float* b2     = (const float*)d_in[5];
  const float* wg     = (const float*)d_in[6];
  const float* bg     = (const float*)d_in[7];
  float* out = (float*)d_out;

  char* w = (char*)d_ws;
  size_t off = 0;
  float*    buf1  = (float*)(w + off);    off += alignup((size_t)NB*CCH*HW*4);
  float*    buf2  = (float*)(w + off);    off += alignup((size_t)NB*CCH*HW*4);
  float*    e_pix = (float*)(w + off);    off += alignup((size_t)NB*CIN*POS*4);
  ushort_t* A1    = (ushort_t*)(w + off); off += alignup((size_t)MROWS*K1PAD*2);
  ushort_t* h_bf  = (ushort_t*)(w + off); off += alignup((size_t)MROWS*K2PAD*2);
  float*    m_f32 = (float*)(w + off);    off += alignup((size_t)NB*D540*POS*4);
  ushort_t* gi    = (ushort_t*)(w + off); off += alignup((size_t)NB*POS*GSTR*2);
  ushort_t* w1f   = (ushort_t*)(w + off); off += alignup((size_t)NT_N*512*2);
  ushort_t* w2f   = (ushort_t*)(w + off); off += alignup((size_t)NT_K2*NT_N*512*2);
  ushort_t* wgf   = (ushort_t*)(w + off); off += alignup((size_t)9*NT_KC*NT_N*512*2);

  const int planes = NB * CCH;

  // Phase 1: SWT levels + quantile-normalized pooling
  swt_level_kernel<<<planes, 256, 0, stream>>>(frames, buf1, e_pix, 1, 0);
  swt_level_kernel<<<planes, 256, 0, stream>>>(buf1,   buf2, e_pix, 2, 1);
  swt_level_kernel<<<planes, 256, 0, stream>>>(buf2,   buf1, e_pix, 4, 2);
  ll_kernel<<<planes, 256, 0, stream>>>(buf1, e_pix);

  // Phase 2: pack operands (bf16, fragment-ordered weights)
  {
    int n1 = NB*D540*POS;
    pack_tclip_kernel<<<(n1 + 255)/256, 256, 0, stream>>>(t_clip, gi);
    int nz = NB*POS*8;
    zerofill_gi_kernel<<<(nz + 255)/256, 256, 0, stream>>>(gi);
    int n2 = NT_N*512;
    pack_w1_kernel<<<(n2 + 255)/256, 256, 0, stream>>>(w1, w1f);
    int n3 = NT_K2*NT_N*512;
    pack_w2_kernel<<<(n3 + 255)/256, 256, 0, stream>>>(w2, w2f);
    int n4 = 9*NT_KC*NT_N*512;
    pack_wg_kernel<<<(n4 + 255)/256, 256, 0, stream>>>(wg, wgf);
    int n5 = MROWS*K1PAD;
    pack_a1_kernel<<<(n5 + 255)/256, 256, 0, stream>>>(e_pix, A1);
  }

  // Phase 3: WMMA GEMMs
  {
    int waves1 = MT_ALL * NT_N;                    // 26656
    gemm1_kernel<<<(waves1 + 7)/8, 256, 0, stream>>>(A1, w1f, b1, h_bf);
    int waves2 = MT_ALL * 9;                       // 7056
    gemm2_kernel<<<(waves2 + 7)/8, 256, 0, stream>>>(h_bf, w2f, b2, m_f32, gi);
  }

  // Phase 4: implicit-GEMM 3x3 conv + sigmoid gate + residual (async-fed)
  conv_kernel<<<NB * MT_IMG, 128, 0, stream>>>(gi, wgf, bg, t_clip, m_f32, out);
}
